// TMHA_64020782514459
// MI455X (gfx1250) — matrix operand order repaired and checked by reading, not compiled, not run on hardware
//
#include <hip/hip_runtime.h>

// ---------------------------------------------------------------------------
// Taylor-attention transformer block for MI455X (gfx1250, wave32, WMMA).
// B=2, S=2048, DIM=1024, H=16, Dh=Dv=64.
// bf16 operands + f32 WMMA accumulation; attention fused (no S*S materialize),
// K/V tiles staged into LDS with async-to-LDS loads (ASYNCcnt), double-buffered.
// Workspace usage: 64 MiB of d_ws.
// ---------------------------------------------------------------------------

#define BATCH 2
#define SEQ   2048
#define DIMM  1024
#define HEADS 16
#define HDIM  64
#define ROWS  (BATCH * SEQ)   // 4096

typedef __attribute__((ext_vector_type(16))) __bf16 v16bf;
typedef __attribute__((ext_vector_type(8)))  float  v8f;

union F16x16 { v16bf v; uint4 q[2]; };

__device__ __forceinline__ unsigned short f2bf(float f) {
  unsigned int u = __builtin_bit_cast(unsigned int, f);
  u += 0x7FFFu + ((u >> 16) & 1u);        // round-to-nearest-even
  return (unsigned short)(u >> 16);
}

__device__ __forceinline__ v8f wmma_bf(v16bf a, v16bf b, v8f c) {
  // v_wmma_f32_16x16x32_bf16: D = A(16x32) * B(32x16) + C(16x16 f32)
  return __builtin_amdgcn_wmma_f32_16x16x32_bf16(
      /*neg_a=*/false, a, /*neg_b=*/false, b,
      /*c_mod=*/(short)0, c, /*reuse_a=*/false, /*reuse_b=*/false);
}

// A-fragment (16x32, MxK): lane<16 holds row (lane&15), K {0..7, 16..23};
// lane>=16 holds same rows with K {8..15, 24..31}. K-contiguous source.
__device__ __forceinline__ v16bf frag_a(const unsigned short* p, int ld,
                                        int row0, int k0) {
  const int lane = threadIdx.x & 31;
  const unsigned short* q =
      p + (size_t)(row0 + (lane & 15)) * ld + k0 + ((lane >> 4) << 3);
  F16x16 f;
  f.q[0] = *(const uint4*)(q);        // K = kb .. kb+7
  f.q[1] = *(const uint4*)(q + 16);   // K = kb+16 .. kb+23
  return f.v;
}

// B-fragment (32x16, KxN) for NT GEMM: B[k,n] = Src[row0+n, k0+k].
// lane<16 holds column n=(lane&15) with K {0..7, 16..23}, lane>=16 with K {8..15, 24..31}.
__device__ __forceinline__ v16bf frag_b(const unsigned short* p, int ld,
                                        int row0, int k0) {
  const int lane = threadIdx.x & 31;
  const unsigned short* q =
      p + (size_t)(row0 + (lane & 15)) * ld + k0 + ((lane >> 4) << 3);
  F16x16 f;
  f.q[0] = *(const uint4*)(q);
  f.q[1] = *(const uint4*)(q + 16);
  return f.v;
}

// --- CDNA5 async-to-LDS path (ASYNCcnt) -----------------------------------
__device__ __forceinline__ unsigned lds_addr(const void* p) {
  // low 32 bits of a generic pointer into LDS = byte offset from LDS base
  return (unsigned)(size_t)p;
}
__device__ __forceinline__ void async_ld16(unsigned lds_off, const void* gp) {
  // per-lane 16B: LDS[lds_off] = MEM[gp]; tracked by ASYNCcnt
  asm volatile("global_load_async_to_lds_b128 %0, %1, off"
               :: "v"(lds_off), "v"(gp) : "memory");
}
__device__ __forceinline__ void wait_async0() {
  asm volatile("s_wait_asynccnt 0x0" ::: "memory");
}

// ---------------------------------------------------------------------------
// f32 -> bf16 conversion
// ---------------------------------------------------------------------------
__global__ void cvt_bf16(const float* __restrict__ x,
                         unsigned short* __restrict__ y, int n) {
  int i = blockIdx.x * blockDim.x + threadIdx.x;
  if (i < n) y[i] = f2bf(x[i]);
}

// ---------------------------------------------------------------------------
// NT GEMM: C[m,n] = scale * sum_k A[m,k] * W[n,k];  M=4096, N=K=1024.
// 256 threads = 8 waves; block tile 128x128; wave tile 32x64 (8 WMMAs/k-step).
// MODE 0: bf16 row-major out.  MODE 1: bf16 transposed per head
// (Vt[(b*H+h)*64+dv][s]).  MODE 2: f32 row-major out.
// ---------------------------------------------------------------------------
template <int MODE>
__global__ __launch_bounds__(256) void gemm_nt(
    const unsigned short* __restrict__ A, const unsigned short* __restrict__ W,
    void* __restrict__ Cp, float scale) {
  const int wid  = threadIdx.x >> 5;
  const int lane = threadIdx.x & 31;
  const int m0 = blockIdx.y * 128 + (wid >> 1) * 32;
  const int n0 = blockIdx.x * 128 + (wid & 1) * 64;

  v8f acc[2][4] = {};
  for (int k0 = 0; k0 < DIMM; k0 += 32) {
    v16bf a0 = frag_a(A, DIMM, m0, k0);
    v16bf a1 = frag_a(A, DIMM, m0 + 16, k0);
    v16bf b0 = frag_b(W, DIMM, n0, k0);
    v16bf b1 = frag_b(W, DIMM, n0 + 16, k0);
    v16bf b2 = frag_b(W, DIMM, n0 + 32, k0);
    v16bf b3 = frag_b(W, DIMM, n0 + 48, k0);
    acc[0][0] = wmma_bf(a0, b0, acc[0][0]);
    acc[0][1] = wmma_bf(a0, b1, acc[0][1]);
    acc[0][2] = wmma_bf(a0, b2, acc[0][2]);
    acc[0][3] = wmma_bf(a0, b3, acc[0][3]);
    acc[1][0] = wmma_bf(a1, b0, acc[1][0]);
    acc[1][1] = wmma_bf(a1, b1, acc[1][1]);
    acc[1][2] = wmma_bf(a1, b2, acc[1][2]);
    acc[1][3] = wmma_bf(a1, b3, acc[1][3]);
  }

  // C/D layout: lane&15 -> n, (lane>>4)*8 + r -> m.
  #pragma unroll
  for (int i = 0; i < 2; i++)
    #pragma unroll
    for (int j = 0; j < 4; j++) {
      const int mi = m0 + i * 16 + ((lane >> 4) << 3);
      const int nj = n0 + j * 16 + (lane & 15);
      if (MODE == 2) {
        float* C = (float*)Cp;
        #pragma unroll
        for (int r = 0; r < 8; r++)
          C[(size_t)(mi + r) * DIMM + nj] = acc[i][j][r] * scale;
      } else if (MODE == 0) {
        unsigned short* C = (unsigned short*)Cp;
        #pragma unroll
        for (int r = 0; r < 8; r++)
          C[(size_t)(mi + r) * DIMM + nj] = f2bf(acc[i][j][r] * scale);
      } else {  // MODE 1: transposed V store, contiguous along s
        unsigned short* C = (unsigned short*)Cp;
        const int b = mi >> 11, s = mi & (SEQ - 1);
        const int h = nj >> 6, dv = nj & 63;
        union { unsigned short s16[8]; uint4 q; } pk;
        #pragma unroll
        for (int r = 0; r < 8; r++) pk.s16[r] = f2bf(acc[i][j][r] * scale);
        *(uint4*)(C + ((size_t)(b * HEADS + h) * HDIM + dv) * SEQ + s) = pk.q;
      }
    }
}

// ---------------------------------------------------------------------------
// Fused Taylor attention. One block = 8 waves = 128 query rows of one (b,h).
// K/V tiles (32 keys) are shared by all 8 waves -> staged cooperatively into
// LDS with async-to-LDS b128 loads (one per thread per tile), double-buffered
// so the next tile's DMA overlaps the current tile's 8 WMMAs.
// ---------------------------------------------------------------------------
__global__ __launch_bounds__(256) void taylor_attn(
    const unsigned short* __restrict__ Qp, const unsigned short* __restrict__ Kp,
    const unsigned short* __restrict__ Vt, const unsigned char* __restrict__ mask,
    unsigned short* __restrict__ Out) {
  __shared__ alignas(16) unsigned short sK[2][32][72];   // [key j][d], padded
  __shared__ alignas(16) unsigned short sV[2][64][40];   // [dv][j],   padded
  __shared__ alignas(16) unsigned short wlds[8][16][40]; // per-wave w tiles

  const int wid  = threadIdx.x >> 5;
  const int lane = threadIdx.x & 31;
  const int bh = blockIdx.x;            // b*16 + h
  const int b = bh >> 4, h = bh & 15;
  const int m0 = blockIdx.y * 128 + wid * 16;  // query row within batch

  const unsigned short* Qb = Qp + (size_t)b * SEQ * DIMM + h * HDIM;
  const unsigned short* Kb = Kp + (size_t)b * SEQ * DIMM + h * HDIM;
  const unsigned short* Vb = Vt + (size_t)bh * HDIM * SEQ;
  const unsigned char* mb = mask + b * SEQ;

  const int c    = lane & 15;           // column (n) index within a tile
  const int mrow = (lane >> 4) << 3;    // row base for this half-wave

  // Per-thread async-copy assignments: K tile 32x64 halfs = 256 x 16B,
  // V tile 64x32 halfs = 256 x 16B.
  const int t   = threadIdx.x;
  const int jj  = t >> 3, ck = (t & 7) * 8;   // K: row j, d-chunk
  const int dvr = t >> 2, cv = (t & 3) * 8;   // V: row dv, j-chunk
  const unsigned short* ksrc = Kb + (size_t)jj * DIMM + ck;
  const unsigned short* vsrc = Vb + (size_t)dvr * SEQ + cv;
  const unsigned kdst[2] = { lds_addr(&sK[0][jj][ck]), lds_addr(&sK[1][jj][ck]) };
  const unsigned vdst[2] = { lds_addr(&sV[0][dvr][cv]), lds_addr(&sV[1][dvr][cv]) };

  auto issue = [&](int j0, int buf) {
    async_ld16(kdst[buf], ksrc + (size_t)j0 * DIMM);
    async_ld16(vdst[buf], vsrc + j0);
  };

  // Q fragments live in registers for the whole key loop (Dh=64 -> 2 frags).
  const v16bf qf0 = frag_a(Qb, DIMM, m0, 0);
  const v16bf qf1 = frag_a(Qb, DIMM, m0, 32);

  v8f   nacc[4] = {};                   // num: 16 q-rows x 64 dv
  float den[8]  = {};                   // per-lane partial row sums

  issue(0, 0);
  for (int i = 0; i < SEQ / 32; i++) {
    const int j0  = i * 32;
    const int buf = i & 1;
    wait_async0();        // own async stores to LDS for buf(i) complete
    __syncthreads();      // everyone's buf(i) visible; buf(i-1) reads done
    if (i + 1 < SEQ / 32) issue(j0 + 32, buf ^ 1);  // overlaps with compute

    const unsigned short* kb_ = &sK[buf][0][0];
    const unsigned short* vb_ = &sV[buf][0][0];

    // s tiles: queries(16) x keys(32), K-dim = Dh = 64 (two k-steps)
    v8f s0 = {}, s1 = {};
    s0 = wmma_bf(qf0, frag_b(kb_, 72, 0, 0),  s0);
    s0 = wmma_bf(qf1, frag_b(kb_, 72, 0, 32), s0);
    s1 = wmma_bf(qf0, frag_b(kb_, 72, 16, 0), s1);
    s1 = wmma_bf(qf1, frag_b(kb_, 72, 16, 32), s1);

    const float mk0 = mb[j0 + c] ? 1.0f : 0.0f;
    const float mk1 = mb[j0 + 16 + c] ? 1.0f : 0.0f;

    #pragma unroll
    for (int r = 0; r < 8; r++) {
      const float w0 = (1.0f + s0[r] + 0.5f * s0[r] * s0[r]) * mk0;
      const float w1 = (1.0f + s1[r] + 0.5f * s1[r] * s1[r]) * mk1;
      den[r] += w0 + w1;
      wlds[wid][mrow + r][c]      = f2bf(w0);   // C-layout -> LDS [m][j]
      wlds[wid][mrow + r][c + 16] = f2bf(w1);
    }

    // Re-read the 16x32 w tile in A-fragment layout (wave-private region,
    // same-wave LDS ordering; compiler inserts the dscnt wait).
    const unsigned short* wp = &wlds[wid][c][0] + ((lane >> 4) << 3);
    F16x16 wf;
    wf.q[0] = *(const uint4*)(wp);
    wf.q[1] = *(const uint4*)(wp + 16);

    // num += w(16x32) * V(32x64) from the LDS-staged, pre-transposed V tile
    #pragma unroll
    for (int tt = 0; tt < 4; tt++)
      nacc[tt] = wmma_bf(wf.v, frag_b(vb_, 40, tt * 16, 0), nacc[tt]);
  }

  // Reduce den across the 16 lanes sharing each query row, invert once.
  #pragma unroll
  for (int r = 0; r < 8; r++) {
    float d = den[r];
    #pragma unroll
    for (int off = 1; off < 16; off <<= 1) d += __shfl_xor(d, off, 32);
    den[r] = 1.0f / d;
  }

  // Out[b*S + m][h*64 + dv] = num/den  (bf16, consumed by the Wo GEMM)
  #pragma unroll
  for (int tt = 0; tt < 4; tt++) {
    const int dv = tt * 16 + c;
    #pragma unroll
    for (int r = 0; r < 8; r++) {
      const int m = m0 + mrow + r;
      Out[((size_t)b * SEQ + m) * DIMM + h * HDIM + dv] =
          f2bf(nacc[tt][r] * den[r]);
    }
  }
}

// ---------------------------------------------------------------------------
// Host-side orchestration (all on `stream`, graph-capture safe).
// Inputs: queries, keys, values (f32), mask (bool/u8), Wq, Wk, Wv, Wo (f32).
// ---------------------------------------------------------------------------
extern "C" void kernel_launch(void* const* d_in, const int* in_sizes, int n_in,
                              void* d_out, int out_size, void* d_ws, size_t ws_size,
                              hipStream_t stream) {
  const float* queries = (const float*)d_in[0];
  const float* keys    = (const float*)d_in[1];
  const float* values  = (const float*)d_in[2];
  const unsigned char* mask = (const unsigned char*)d_in[3];  // bool -> 1 byte
  const float* Wq = (const float*)d_in[4];
  const float* Wk = (const float*)d_in[5];
  const float* Wv = (const float*)d_in[6];
  const float* Wo = (const float*)d_in[7];

  const size_t MB = 1u << 20;
  char* ws = (char*)d_ws;   // needs 64 MiB
  unsigned short* qbf = (unsigned short*)(ws + 0 * MB);
  unsigned short* kbf = (unsigned short*)(ws + 8 * MB);
  unsigned short* vbf = (unsigned short*)(ws + 16 * MB);
  unsigned short* wqb = (unsigned short*)(ws + 24 * MB);
  unsigned short* wkb = (unsigned short*)(ws + 26 * MB);
  unsigned short* wvb = (unsigned short*)(ws + 28 * MB);
  unsigned short* wob = (unsigned short*)(ws + 30 * MB);
  unsigned short* Qp  = (unsigned short*)(ws + 32 * MB);
  unsigned short* Kp  = (unsigned short*)(ws + 40 * MB);
  unsigned short* Vtp = (unsigned short*)(ws + 48 * MB);
  unsigned short* At  = (unsigned short*)(ws + 56 * MB);

  const int nAct = ROWS * DIMM;   // 4096*1024
  const int nWgt = DIMM * DIMM;   // 1024*1024
  cvt_bf16<<<nAct / 256, 256, 0, stream>>>(queries, qbf, nAct);
  cvt_bf16<<<nAct / 256, 256, 0, stream>>>(keys,    kbf, nAct);
  cvt_bf16<<<nAct / 256, 256, 0, stream>>>(values,  vbf, nAct);
  cvt_bf16<<<nWgt / 256, 256, 0, stream>>>(Wq, wqb, nWgt);
  cvt_bf16<<<nWgt / 256, 256, 0, stream>>>(Wk, wkb, nWgt);
  cvt_bf16<<<nWgt / 256, 256, 0, stream>>>(Wv, wvb, nWgt);
  cvt_bf16<<<nWgt / 256, 256, 0, stream>>>(Wo, wob, nWgt);

  dim3 ggrid(DIMM / 128, ROWS / 128);  // (8, 32)
  // Q projection with folded scale Dh^-0.5 = 0.125 (exact power of two)
  gemm_nt<0><<<ggrid, 256, 0, stream>>>(qbf, wqb, (void*)Qp, 0.125f);
  gemm_nt<0><<<ggrid, 256, 0, stream>>>(kbf, wkb, (void*)Kp, 1.0f);
  gemm_nt<1><<<ggrid, 256, 0, stream>>>(vbf, wvb, (void*)Vtp, 1.0f);  // transposed

  dim3 agrid(BATCH * HEADS, SEQ / 128);  // (32, 16)
  taylor_attn<<<agrid, 256, 0, stream>>>(Qp, Kp, Vtp, mask, At);

  gemm_nt<2><<<ggrid, 256, 0, stream>>>(At, wob, d_out, 1.0f);  // f32 out
}
